// ScaleDotProductAttention_14680198218247
// MI455X (gfx1250) — matrix operand order repaired and checked by reading, not compiled, not run on hardware
//
#include <hip/hip_runtime.h>
#include <hip/hip_bf16.h>

#define B_  16
#define N_  4096
#define D_  128
#define QB  128          // query rows per workgroup
#define KT  32           // keys per tile
#define NW  8            // waves per workgroup
#define NT  (N_ / KT)    // 128 key tiles
#define SCALE 0.08838834764831845f   // 1/sqrt(128)

typedef __attribute__((ext_vector_type(16))) __bf16  v16bf;
typedef __attribute__((ext_vector_type(8)))  float   v8f;
typedef __attribute__((ext_vector_type(4)))  unsigned int u32x4;
typedef __attribute__((ext_vector_type(8)))  int     i32x8;
typedef __attribute__((ext_vector_type(4)))  int     i32x4;
typedef __attribute__((ext_vector_type(4)))  float   f32x4;

struct FragBits { u32x4 lo, hi; };

static __device__ inline v16bf make_frag(u32x4 lo, u32x4 hi) {
    FragBits t{lo, hi};
    return __builtin_bit_cast(v16bf, t);
}

static __device__ inline unsigned short f2bf(float f) {
    unsigned u = __builtin_bit_cast(unsigned, f);
    u += 0x7FFFu + ((u >> 16) & 1u);          // round-to-nearest-even
    return (unsigned short)(u >> 16);
}

// ---------------------------------------------------------------------------
// TDM: DMA one 32x128 bf16 tile (row-major) from global into LDS.
// D# per CDNA5 ISA ch.8: group0 = {count/ctrl, lds_addr, global_addr, type=2},
// group1 = {data_size, tensor dims, tile dims, strides}. 2D tensor -> groups
// 2/3 zero. Wave-level op tracked by TENSORcnt.
// ---------------------------------------------------------------------------
static __device__ inline void tdm_load_tile(const unsigned short* gsrc, unsigned lds_byte_addr) {
    unsigned long long ga = (unsigned long long)(uintptr_t)gsrc;
    u32x4 g0;
    g0.x = 1u;                                              // count=1, user descriptor
    g0.y = lds_byte_addr;                                   // lds_addr[31:0]
    g0.z = (unsigned)(ga & 0xffffffffu);                    // global_addr[31:0]
    g0.w = (unsigned)((ga >> 32) & 0x01ffffffu) | (2u << 30); // global_addr[56:32] | type=2
    i32x8 g1;
    g1[0] = 0x00010000;                                     // data_size=1 (2 bytes/elem)
    g1[1] = (int)((D_ & 0xffff) << 16);                     // tensor_dim0[15:0]=128
    g1[2] = (int)((N_ & 0xffff) << 16);                     // tensor_dim0 hi=0 | tensor_dim1 lo=4096
    g1[3] = (int)(D_ << 16);                                // tensor_dim1 hi=0 | tile_dim0=128
    g1[4] = KT;                                             // tile_dim1=32, tile_dim2=0
    g1[5] = D_;                                             // tensor_dim0_stride=128
    g1[6] = 0;
    g1[7] = 0;
    i32x4 z4 = {0, 0, 0, 0};
#if __clang_major__ >= 23
    i32x8 z8 = {0, 0, 0, 0, 0, 0, 0, 0};
    __builtin_amdgcn_tensor_load_to_lds(g0, g1, z4, z4, z8, 0);
#else
    __builtin_amdgcn_tensor_load_to_lds(g0, g1, z4, z4, 0);
#endif
}

// ---------------------------------------------------------------------------
// fp32 -> bf16 conversion (bandwidth-bound preprocessing pass)
// ---------------------------------------------------------------------------
__global__ __launch_bounds__(256)
void cvt_f32_bf16(const float* __restrict__ in, unsigned short* __restrict__ out, int n) {
    int i = (blockIdx.x * 256 + threadIdx.x) * 8;
    if (i >= n) return;
    f32x4 a = *(const f32x4*)(in + i);
    f32x4 b = *(const f32x4*)(in + i + 4);
    unsigned short tmp[8];
    tmp[0] = f2bf(a.x); tmp[1] = f2bf(a.y); tmp[2] = f2bf(a.z); tmp[3] = f2bf(a.w);
    tmp[4] = f2bf(b.x); tmp[5] = f2bf(b.y); tmp[6] = f2bf(b.z); tmp[7] = f2bf(b.w);
    *(u32x4*)(out + i) = *(u32x4*)tmp;
}

// ---------------------------------------------------------------------------
// Flash attention forward, bf16 WMMA, f32 accumulation, TDM-staged K tiles
// ---------------------------------------------------------------------------
__global__ __launch_bounds__(256)
void fa_fwd(const unsigned short* __restrict__ qg,
            const unsigned short* __restrict__ kg,
            const unsigned short* __restrict__ vg,
            float* __restrict__ outp) {
    __shared__ __align__(16) unsigned short Ks[2][KT][D_];   // 16 KB, row-major [key][dim] (TDM dest)
    __shared__ __align__(16) unsigned short Vs[2][D_][KT];   // 16 KB, transposed [dim][key]
    __shared__ __align__(16) unsigned short Ps[NW][16][KT];  //  8 KB, per-wave P scratch

    const int tid = threadIdx.x;
    const int w   = tid >> 5;          // wave id 0..7
    const int l   = tid & 31;          // lane
    const int lh  = l >> 4;            // lane half (0/1)
    const int ln  = l & 15;
    const int b   = blockIdx.y;
    const int qr  = blockIdx.x * QB + w * 16;   // wave's first query row

    const unsigned short* Qg = qg + ((size_t)b * N_ + qr) * D_;
    const unsigned short* Kg = kg + (size_t)b * N_ * D_;
    const unsigned short* Vg = vg + (size_t)b * N_ * D_;

    // ---- load Q A-fragments (16x32 per 32-dim chunk, 4 chunks over D=128) ----
    // A layout: lanes 0-15: M=lane, K=0..15; lanes 16-31: K=16..31
    v16bf qf[4];
#pragma unroll
    for (int c = 0; c < 4; ++c) {
        const unsigned short* p = Qg + ln * D_ + c * 32 + lh * 16;
        qf[c] = make_frag(*(const u32x4*)p, *(const u32x4*)(p + 8));
    }

    const v8f vzero = {0.f, 0.f, 0.f, 0.f, 0.f, 0.f, 0.f, 0.f};
    v8f acc[8];
#pragma unroll
    for (int t = 0; t < 8; ++t) acc[t] = vzero;
    float mrow[8], lrow[8];
#pragma unroll
    for (int i = 0; i < 8; ++i) { mrow[i] = -1e30f; lrow[i] = 0.f; }

    // ---- wave0 issues TDM for K tile; all threads transpose-stage V tile ----
    auto stageK = [&](int j, int buf) {   // call from wave 0 only
        tdm_load_tile(Kg + (size_t)j * KT * D_,
                      (unsigned)(uintptr_t)(&Ks[buf][0][0]));
    };
    auto stageV = [&](int j, int buf) {
        const unsigned short* src = Vg + (size_t)j * KT * D_ + tid * 16;
        u32x4 a = *(const u32x4*)src;
        u32x4 c = *(const u32x4*)(src + 8);
        unsigned short tmp[16];
        *(u32x4*)tmp       = a;
        *(u32x4*)(tmp + 8) = c;
        const int e  = tid * 16;
        const int r  = e / D_;     // local key row 0..31
        const int c0 = e % D_;     // dim column (multiple of 16)
#pragma unroll
        for (int k2 = 0; k2 < 16; ++k2)
            Vs[buf][c0 + k2][r] = tmp[k2];
    };

    if (w == 0) stageK(0, 0);
    stageV(0, 0);
    if (w == 0) __builtin_amdgcn_s_wait_tensorcnt(0);
    __syncthreads();

    for (int j = 0; j < NT; ++j) {
        const int buf = j & 1;
        if (j + 1 < NT) {
            if (w == 0) stageK(j + 1, buf ^ 1);   // async DMA overlaps compute below
            stageV(j + 1, buf ^ 1);
        }
        if (j + 2 < NT)
            __builtin_prefetch(Vg + (size_t)(j + 2) * KT * D_, 0, 0);

        // ---- S(16x32) = Q(16x128) * K^T : preload all 8 B-frags, then 8 wmma ----
        // B layout: lane: N=ln (key), K = lh*16..+15 (dims) -> contiguous in Ks row
        v16bf kb0[4], kb1[4];
#pragma unroll
        for (int c = 0; c < 4; ++c) {
            const unsigned short* kp0 = &Ks[buf][ln][c * 32 + lh * 16];
            const unsigned short* kp1 = &Ks[buf][16 + ln][c * 32 + lh * 16];
            kb0[c] = make_frag(*(const u32x4*)kp0, *(const u32x4*)(kp0 + 8));
            kb1[c] = make_frag(*(const u32x4*)kp1, *(const u32x4*)(kp1 + 8));
        }
        v8f s0 = vzero, s1 = vzero;
#pragma unroll
        for (int c = 0; c < 4; ++c) {
            s0 = __builtin_amdgcn_wmma_f32_16x16x32_bf16(false, qf[c], false, kb0[c], (short)0, s0, false, false);
            s1 = __builtin_amdgcn_wmma_f32_16x16x32_bf16(false, qf[c], false, kb1[c], (short)0, s1, false, false);
        }

        // ---- online softmax update ----
        float alpha[8];
#pragma unroll
        for (int i = 0; i < 8; ++i) {
            float t = fmaxf(s0[i], s1[i]);
            t = fmaxf(t, __shfl_xor(t, 1, 32));
            t = fmaxf(t, __shfl_xor(t, 2, 32));
            t = fmaxf(t, __shfl_xor(t, 4, 32));
            t = fmaxf(t, __shfl_xor(t, 8, 32));   // row max within 16-lane half
            float mnew = fmaxf(mrow[i], t);
            alpha[i] = __expf((mrow[i] - mnew) * SCALE);
            mrow[i] = mnew;
            float p0 = __expf((s0[i] - mnew) * SCALE);
            float p1 = __expf((s1[i] - mnew) * SCALE);
            float rs = p0 + p1;
            rs += __shfl_xor(rs, 1, 32);
            rs += __shfl_xor(rs, 2, 32);
            rs += __shfl_xor(rs, 4, 32);
            rs += __shfl_xor(rs, 8, 32);
            lrow[i] = lrow[i] * alpha[i] + rs;
            const int row = lh * 8 + i;           // C layout: lane<16 -> M=i, else M=8+i
            Ps[w][row][ln]      = f2bf(p0);
            Ps[w][row][16 + ln] = f2bf(p1);
        }
#pragma unroll
        for (int t = 0; t < 8; ++t) {
#pragma unroll
            for (int i = 0; i < 8; ++i) acc[t][i] *= alpha[i];
        }

        // ---- re-layout P: C-tile -> A-fragment via same-wave LDS (in-order) ----
        const unsigned short* pp = &Ps[w][ln][lh * 16];
        v16bf pf = make_frag(*(const u32x4*)pp, *(const u32x4*)(pp + 8));

        // ---- O(16x128) += P(16x32) * V(32x128): preload 8 B-frags, 8 wmma ----
        v16bf vb[8];
#pragma unroll
        for (int t = 0; t < 8; ++t) {
            const unsigned short* vp = &Vs[buf][t * 16 + ln][lh * 16];
            vb[t] = make_frag(*(const u32x4*)vp, *(const u32x4*)(vp + 8));
        }
#pragma unroll
        for (int t = 0; t < 8; ++t)
            acc[t] = __builtin_amdgcn_wmma_f32_16x16x32_bf16(false, pf, false, vb[t], (short)0, acc[t], false, false);

        if (w == 0) __builtin_amdgcn_s_wait_tensorcnt(0);  // K DMA for next tile done
        __syncthreads();
    }

    // ---- finalize: O /= l, scatter fp32 output ----
    float rinv[8];
#pragma unroll
    for (int i = 0; i < 8; ++i) rinv[i] = 1.0f / lrow[i];
    float* Og = outp + ((size_t)b * N_ + qr) * D_;
#pragma unroll
    for (int t = 0; t < 8; ++t) {
#pragma unroll
        for (int i = 0; i < 8; ++i)
            Og[(lh * 8 + i) * D_ + t * 16 + ln] = acc[t][i] * rinv[i];
    }
}

// ---------------------------------------------------------------------------
extern "C" void kernel_launch(void* const* d_in, const int* in_sizes, int n_in,
                              void* d_out, int out_size, void* d_ws, size_t ws_size,
                              hipStream_t stream) {
    (void)in_sizes; (void)n_in; (void)out_size; (void)ws_size;
    const float* q = (const float*)d_in[0];
    const float* k = (const float*)d_in[1];
    const float* v = (const float*)d_in[2];

    const size_t elems = (size_t)B_ * N_ * D_;       // 8,388,608 per tensor
    unsigned short* qb = (unsigned short*)d_ws;
    unsigned short* kb = qb + elems;
    unsigned short* vb = kb + elems;

    const int n = (int)elems;
    const int blocks = n / (8 * 256);                // exact: 4096
    cvt_f32_bf16<<<blocks, 256, 0, stream>>>(q, qb, n);
    cvt_f32_bf16<<<blocks, 256, 0, stream>>>(k, kb, n);
    cvt_f32_bf16<<<blocks, 256, 0, stream>>>(v, vb, n);

    dim3 grid(N_ / QB, B_);                          // (32, 16)
    fa_fwd<<<grid, 256, 0, stream>>>(qb, kb, vb, (float*)d_out);
}
